// SelfAttentionLayer_44908178047403
// MI455X (gfx1250) — matrix operand order repaired and checked by reading, not compiled, not run on hardware
//
#include <hip/hip_runtime.h>
#include <stdint.h>

// ---------------------------------------------------------------- constants
#define B_   4
#define T_   2048
#define D_   1024
#define TD3  (3 * D_)
#define MTOT (B_ * T_)       // 8192 rows of x

// GEMM block tile
#define BM   128
#define BN   128
#define BK   64
#define PAD  8
#define LDK  (BK + PAD)      // 72 elements -> 144B row stride in LDS

// ---------------------------------------------------------------- types
typedef __attribute__((ext_vector_type(16))) __bf16        v16bf;
typedef __attribute__((ext_vector_type(8)))  float         v8f;
typedef __attribute__((ext_vector_type(4)))  unsigned int  v4u;
typedef __attribute__((ext_vector_type(8)))  unsigned int  v8u;

// 16-byte GCC-style int vector: element type expected by the async-LDS builtin.
typedef int v4i_b128 __attribute__((vector_size(16)));
typedef __attribute__((address_space(1))) v4i_b128* g_b128_ptr;   // global
typedef __attribute__((address_space(3))) v4i_b128* l_b128_ptr;   // LDS

union Pun {
    v16bf bf;
    v8u   u;
    struct { v4u lo; v4u hi; } s;
};

// ---------------------------------------------------------------- helpers
__device__ __forceinline__ unsigned short f2bf(float f) {
    unsigned u = __float_as_uint(f);
    u += 0x7FFFu + ((u >> 16) & 1u);        // round-to-nearest-even
    return (unsigned short)(u >> 16);
}
__device__ __forceinline__ float bf2f(unsigned short h) {
    return __uint_as_float(((unsigned)h) << 16);
}

// ---------------- async global->LDS copy (16B per lane), ASYNCcnt-tracked
#if __has_builtin(__builtin_amdgcn_global_load_async_to_lds_b128)
#define ASYNC_LDS 1
#else
#define ASYNC_LDS 0
#endif

__device__ __forceinline__ void cp16(const unsigned short* g, unsigned short* l) {
#if ASYNC_LDS
    __builtin_amdgcn_global_load_async_to_lds_b128(
        (g_b128_ptr)(uintptr_t)g,
        (l_b128_ptr)(uint32_t)(uintptr_t)l,
        0, 0);
#else
    *(v4u*)l = *(const v4u*)g;              // fallback: sync copy through VGPRs
#endif
}
__device__ __forceinline__ void wait_async() {
#if ASYNC_LDS
#if __has_builtin(__builtin_amdgcn_s_wait_asynccnt)
    __builtin_amdgcn_s_wait_asynccnt(0);
#else
    asm volatile("s_wait_asynccnt 0x0" ::: "memory");
#endif
#endif
}

// A operand (16x32 bf16, MxK) from LDS. lane m = lane&15, half = lane>>4.
// element e -> K = e + 8*half + (e>=8 ? 8 : 0)  => two contiguous 16B chunks.
__device__ __forceinline__ v16bf load_a(const unsigned short* p, int half) {
    Pun t;
    t.s.lo = *(const v4u*)(p + 8 * half);
    t.s.hi = *(const v4u*)(p + 16 + 8 * half);
    return t.bf;
}
// B operand (32x16 bf16, KxN) sourced from row-major Bw[N][K] (i.e. B = Bw^T).
// lane n = lane&15, half = lane>>4. element e -> K = e + 8*half + (e>=8 ? 8 : 0)  => two 16B chunks.
__device__ __forceinline__ v16bf load_b(const unsigned short* p, int half) {
    Pun t;
    t.s.lo = *(const v4u*)(p + 8 * half);
    t.s.hi = *(const v4u*)(p + 16 + 8 * half);
    return t.bf;
}

// Per-wave compute on one staged LDS panel: 32x64 strip (8 accumulators), K = BK.
// Each B fragment feeds two WMMAs (rows +0 and +16) -> 1.5 ds_load per wmma.
__device__ __forceinline__ void tile_compute(const unsigned short* As_,
                                             const unsigned short* Bs_,
                                             int wm, int wn, int lane, v8f cs[8]) {
    const int half = lane >> 4, l15 = lane & 15;
    const unsigned short* arow0 = As_ + (wm * 32 + l15) * LDK;
    const unsigned short* arow1 = arow0 + 16 * LDK;
    const unsigned short* brow  = Bs_ + (wn * 64 + l15) * LDK;
    #pragma unroll
    for (int kk = 0; kk < BK; kk += 32) {
        v16bf a0 = load_a(arow0 + kk, half);
        v16bf a1 = load_a(arow1 + kk, half);
        #pragma unroll
        for (int j = 0; j < 4; ++j) {
            v16bf b = load_b(brow + j * 16 * LDK + kk, half);
            cs[j] = __builtin_amdgcn_wmma_f32_16x16x32_bf16(
                false, a0, false, b, (short)0, cs[j], false, false);
            cs[4 + j] = __builtin_amdgcn_wmma_f32_16x16x32_bf16(
                false, a1, false, b, (short)0, cs[4 + j], false, false);
        }
    }
}

// Block-level GEMM: C(BM x BN) = A[m0.., k] * Bw[n0.., k]^T over K = [0, kEnd).
// Double-buffered LDS (72 KB of the WGP's 320 KB) fed by async global->LDS copies.
__device__ __forceinline__ void tile_gemm(const unsigned short* __restrict__ A, int lda,
                                          const unsigned short* __restrict__ Bw, int ldb,
                                          int m0, int n0, int kEnd, v8f cs[8]) {
    __shared__ unsigned short As[2][BM * LDK];   // 2 * 18 KB
    __shared__ unsigned short Bs[2][BN * LDK];   // 2 * 18 KB
    const int tid  = threadIdx.x;
    const int lane = tid & 31, wave = tid >> 5;
    const int wm = wave >> 1, wn = wave & 1;

    auto stage = [&](int kt, int buf) {
        const unsigned short* Ag = A + (long)m0 * lda + kt * BK;
        const unsigned short* Bg = Bw + (long)n0 * ldb + kt * BK;
        #pragma unroll
        for (int it = 0; it < 4; ++it) {                 // 1024 x 16B chunks of A
            int c = tid + 256 * it;
            int row = c >> 3, col = (c & 7) * 8;
            cp16(Ag + (long)row * lda + col, &As[buf][row * LDK + col]);
        }
        #pragma unroll
        for (int it = 0; it < 4; ++it) {                 // 1024 x 16B chunks of B
            int c = tid + 256 * it;
            int row = c >> 3, col = (c & 7) * 8;
            cp16(Bg + (long)row * ldb + col, &Bs[buf][row * LDK + col]);
        }
    };

    const int nk = kEnd / BK;
    int buf = 0;
    stage(0, 0);
    wait_async();
    __syncthreads();
    for (int kt = 0; kt < nk; ++kt) {
        if (kt + 1 < nk) stage(kt + 1, buf ^ 1);         // overlap with compute
        tile_compute(&As[buf][0], &Bs[buf][0], wm, wn, lane, cs);
        wait_async();
        __syncthreads();
        buf ^= 1;
    }
}

// ---------------------------------------------------------------- kernels
__global__ void k_f2bf(const float* __restrict__ in, unsigned short* __restrict__ out, int n) {
    int i = blockIdx.x * 256 + threadIdx.x;
    if (i < n) out[i] = f2bf(in[i]);
}

// kqv = x @ W_kqv^T + b ; split (k,q,v); v stored transposed [B][D][T] in bf16.
__global__ void k_gemm_kqv(const unsigned short* __restrict__ xbf,
                           const unsigned short* __restrict__ wbf,
                           const float* __restrict__ bias,
                           unsigned short* __restrict__ qb,
                           unsigned short* __restrict__ kb,
                           unsigned short* __restrict__ vT) {
    const int n0 = blockIdx.x * BN;
    const int m0 = blockIdx.y * BM;
    v8f cs[8] = {};
    tile_gemm(xbf, D_, wbf, D_, m0, n0, D_, cs);

    const int lane = threadIdx.x & 31, wave = threadIdx.x >> 5;
    const int half = lane >> 4, nl = lane & 15;
    const int mwB = m0 + (wave >> 1) * 32;
    const int nw  = n0 + (wave & 1) * 64;
    #pragma unroll
    for (int mi = 0; mi < 2; ++mi) {
        const int mw = mwB + mi * 16;
        #pragma unroll
        for (int j = 0; j < 4; ++j) {
            const int e  = nw + 16 * j + nl;
            const float bv = bias[e];
            #pragma unroll
            for (int r = 0; r < 8; ++r) {
                const int m = mw + 8 * half + r;
                const unsigned short val = f2bf(cs[mi * 4 + j][r] + bv);
                if (e < D_) {
                    kb[(long)m * D_ + e] = val;                       // split order: k first
                } else if (e < 2 * D_) {
                    qb[(long)m * D_ + (e - D_)] = val;
                } else {
                    const int b = m / T_, t = m % T_;
                    vT[((long)b * D_ + (e - 2 * D_)) * T_ + t] = val; // transposed V
                }
            }
        }
    }
}

// S[b,t,s] = q[b,t,:] . k[b,s,:]  (unscaled; scale applied in softmax).
__global__ void k_gemm_scores(const unsigned short* __restrict__ qb,
                              const unsigned short* __restrict__ kb,
                              unsigned short* __restrict__ S) {
    const int b  = blockIdx.z;
    const int n0 = blockIdx.x * BN;                  // key index s
    const int m0 = blockIdx.y * BM;                  // query index t
    if (n0 > m0 + BM - 1) return;                    // block fully above diagonal

    const unsigned short* q = qb + (long)b * T_ * D_;
    const unsigned short* k = kb + (long)b * T_ * D_;
    v8f cs[8] = {};
    tile_gemm(q, D_, k, D_, m0, n0, D_, cs);

    unsigned short* Sp = S + (long)b * T_ * T_;
    const int lane = threadIdx.x & 31, wave = threadIdx.x >> 5;
    const int half = lane >> 4, nl = lane & 15;
    const int mwB = m0 + (wave >> 1) * 32;
    const int nw  = n0 + (wave & 1) * 64;
    #pragma unroll
    for (int mi = 0; mi < 2; ++mi)
        #pragma unroll
        for (int j = 0; j < 4; ++j)
            #pragma unroll
            for (int r = 0; r < 8; ++r) {
                const int t = mwB + mi * 16 + 8 * half + r;
                Sp[(long)t * T_ + (nw + 16 * j + nl)] = f2bf(cs[mi * 4 + j][r]);
            }
}

// Row-wise causal softmax, in place (S -> P), one 256-thread block per row.
__global__ void k_softmax(unsigned short* __restrict__ S) {
    __shared__ float red[256];
    const int row = blockIdx.x;                    // 0 .. B*T-1
    const int b   = row >> 11;                     // T_ == 2048
    const int t   = row & (T_ - 1);
    unsigned short* Sp = S + ((long)b * T_ + t) * T_;
    const float scale = 0.022097086912079608f;     // 1/sqrt(2048)
    const int len = t + 1;
    const int tid = threadIdx.x;

    float mx = -3.0e38f;
    for (int i = tid; i < len; i += 256) mx = fmaxf(mx, scale * bf2f(Sp[i]));
    red[tid] = mx; __syncthreads();
    for (int s = 128; s > 0; s >>= 1) { if (tid < s) red[tid] = fmaxf(red[tid], red[tid + s]); __syncthreads(); }
    mx = red[0]; __syncthreads();

    float sum = 0.f;
    for (int i = tid; i < len; i += 256) sum += __expf(scale * bf2f(Sp[i]) - mx);
    red[tid] = sum; __syncthreads();
    for (int s = 128; s > 0; s >>= 1) { if (tid < s) red[tid] += red[tid + s]; __syncthreads(); }
    const float inv = 1.0f / red[0];

    for (int i = tid; i < T_; i += 256) {
        const float p = (i < len) ? __expf(scale * bf2f(Sp[i]) - mx) * inv : 0.0f;
        Sp[i] = f2bf(p);
    }
}

// O[b,t,d] = sum_s P[b,t,s] * V[b,s,d]  using vT[b,d,s]; K-loop trimmed by causality.
__global__ void k_gemm_pv(const unsigned short* __restrict__ P,
                          const unsigned short* __restrict__ vT,
                          unsigned short* __restrict__ Ob) {
    const int b  = blockIdx.z;
    const int n0 = blockIdx.x * BN;                  // d
    const int m0 = blockIdx.y * BM;                  // t
    const unsigned short* Pp = P + (long)b * T_ * T_;
    const unsigned short* Vp = vT + (long)b * D_ * T_;
    v8f cs[8] = {};
    tile_gemm(Pp, T_, Vp, T_, m0, n0, m0 + BM, cs);  // P[t,s]=0 for s>t

    const int lane = threadIdx.x & 31, wave = threadIdx.x >> 5;
    const int half = lane >> 4, nl = lane & 15;
    const int mwB = m0 + (wave >> 1) * 32;
    const int nw  = n0 + (wave & 1) * 64;
    #pragma unroll
    for (int mi = 0; mi < 2; ++mi)
        #pragma unroll
        for (int j = 0; j < 4; ++j)
            #pragma unroll
            for (int r = 0; r < 8; ++r) {
                const int t = mwB + mi * 16 + 8 * half + r;
                Ob[((long)b * T_ + t) * D_ + (nw + 16 * j + nl)] = f2bf(cs[mi * 4 + j][r]);
            }
}

// out = O @ W_proj^T + b_proj (fp32 output).
__global__ void k_gemm_proj(const unsigned short* __restrict__ Ob,
                            const unsigned short* __restrict__ wp,
                            const float* __restrict__ bias,
                            float* __restrict__ out) {
    const int n0 = blockIdx.x * BN;
    const int m0 = blockIdx.y * BM;
    v8f cs[8] = {};
    tile_gemm(Ob, D_, wp, D_, m0, n0, D_, cs);

    const int lane = threadIdx.x & 31, wave = threadIdx.x >> 5;
    const int half = lane >> 4, nl = lane & 15;
    const int mwB = m0 + (wave >> 1) * 32;
    const int nw  = n0 + (wave & 1) * 64;
    #pragma unroll
    for (int mi = 0; mi < 2; ++mi)
        #pragma unroll
        for (int j = 0; j < 4; ++j) {
            const int e = nw + 16 * j + nl;
            const float bv = bias[e];
            #pragma unroll
            for (int r = 0; r < 8; ++r) {
                const int m = mwB + mi * 16 + 8 * half + r;
                out[(long)m * D_ + e] = cs[mi * 4 + j][r] + bv;
            }
        }
}

// ---------------------------------------------------------------- launch
extern "C" void kernel_launch(void* const* d_in, const int* in_sizes, int n_in,
                              void* d_out, int out_size, void* d_ws, size_t ws_size,
                              hipStream_t stream) {
    (void)in_sizes; (void)n_in; (void)out_size; (void)ws_size;
    const float* x  = (const float*)d_in[0];
    const float* Wk = (const float*)d_in[1];
    const float* bk = (const float*)d_in[2];
    const float* Wp = (const float*)d_in[3];
    const float* bp = (const float*)d_in[4];
    float* out = (float*)d_out;

    char* ws = (char*)d_ws;
    size_t off = 0;
    auto alloc = [&](size_t bytes) -> void* {
        void* p = ws + off;
        off += (bytes + 255) & ~(size_t)255;
        return p;
    };
    unsigned short* xbf  = (unsigned short*)alloc((size_t)MTOT * D_ * 2);    // 16 MB
    unsigned short* wkbf = (unsigned short*)alloc((size_t)TD3 * D_ * 2);     //  6 MB
    unsigned short* wpbf = (unsigned short*)alloc((size_t)D_ * D_ * 2);      //  2 MB
    unsigned short* qb   = (unsigned short*)alloc((size_t)MTOT * D_ * 2);    // 16 MB
    unsigned short* kb   = (unsigned short*)alloc((size_t)MTOT * D_ * 2);    // 16 MB
    unsigned short* vT   = (unsigned short*)alloc((size_t)MTOT * D_ * 2);    // 16 MB
    unsigned short* S    = (unsigned short*)alloc((size_t)B_ * T_ * T_ * 2); // 32 MB
    unsigned short* Ob   = (unsigned short*)alloc((size_t)MTOT * D_ * 2);    // 16 MB

    dim3 blk(256, 1, 1);
    k_f2bf<<<(MTOT * D_ + 255) / 256, blk, 0, stream>>>(x, xbf, MTOT * D_);
    k_f2bf<<<(TD3 * D_ + 255) / 256, blk, 0, stream>>>(Wk, wkbf, TD3 * D_);
    k_f2bf<<<(D_ * D_ + 255) / 256, blk, 0, stream>>>(Wp, wpbf, D_ * D_);

    k_gemm_kqv   <<<dim3(TD3 / BN, MTOT / BM, 1),  blk, 0, stream>>>(xbf, wkbf, bk, qb, kb, vT);
    k_gemm_scores<<<dim3(T_ / BN,  T_ / BM,   B_), blk, 0, stream>>>(qb, kb, S);
    k_softmax    <<<dim3(MTOT, 1, 1),              blk, 0, stream>>>(S);
    k_gemm_pv    <<<dim3(D_ / BN,  T_ / BM,   B_), blk, 0, stream>>>(S, vT, Ob);
    k_gemm_proj  <<<dim3(D_ / BN,  MTOT / BM, 1),  blk, 0, stream>>>(Ob, wpbf, bp, out);
}
